// GINNet_70274254897753
// MI455X (gfx1250) — matrix operand order repaired and checked by reading, not compiled, not run on hardware
//
#include <hip/hip_runtime.h>
#include <stdint.h>

typedef __attribute__((ext_vector_type(16))) _Float16 v16h;
typedef __attribute__((ext_vector_type(8)))  float    v8f;

#define N_CLASS 10
#define HIDDEN  128
#define BN_EPS  1e-5f

// ---------------------------------------------------------------- utilities
__global__ void zero_f32(float* __restrict__ p, int n) {
  int i = blockIdx.x * blockDim.x + threadIdx.x;
  if (i < n) p[i] = 0.0f;
}

__global__ void copy_f32(const float* __restrict__ in, float* __restrict__ out, long n) {
  long i = (long)blockIdx.x * blockDim.x + threadIdx.x;
  if (i < n) out[i] = in[i];
}

// H[dst] += X[src], one thread per (edge, feature); H pre-initialized with X
template <int D>
__global__ void scatter_add(const float* __restrict__ X, const int* __restrict__ src,
                            const int* __restrict__ dst, float* __restrict__ H, long total) {
  long i = (long)blockIdx.x * blockDim.x + threadIdx.x;
  if (i >= total) return;
  long e = i / D;
  int  k = (int)(i - e * D);
  int  s = src[e];
  int  d = dst[e];
  atomicAdd(&H[(long)d * D + k], X[(long)s * D + k]);
}

// ------------------------------------------------- pack weights to B-fragments
// B fragment (16x16x32 f16 WMMA): lane (l&15)=column, (l>>4) selects K 0..15 vs
// 16..31 within the 32-deep k-tile; 16 contiguous halves per lane per fragment.
__global__ void pack_weights(const float* __restrict__ W, _Float16* __restrict__ dstp,
                             int ktiles, int ntiles, int Kreal, int Ncreal) {
  int idx = blockIdx.x * blockDim.x + threadIdx.x;
  int total = ktiles * ntiles * 32;
  if (idx >= total) return;
  int fid  = idx >> 5;
  int lane = idx & 31;
  int kt   = fid / ntiles;
  int nt   = fid - kt * ntiles;
  int col  = nt * 16 + (lane & 15);
  int kb   = kt * 32 + (lane >> 4) * 16;
  _Float16* o = dstp + ((size_t)fid * 32 + lane) * 16;
  for (int j = 0; j < 16; ++j) {
    int k = kb + j;
    float v = (k < Kreal && col < Ncreal) ? W[(size_t)k * Ncreal + col] : 0.0f;
    o[j] = (_Float16)v;
  }
}

// ------------------------------------------------------------- WMMA GEMM
// Out[M,NCPAD] = A[M,KREAL] @ W + bias (cols >= NCREAL hold garbage, never read)
// per-column sum/sumsq fused for BN. grid.x = M/16 tiles; blockDim = 32*NTILES.
// All shapes compile-time: full unroll, unconditional branch-free stores.
template <int NKT, int NTILES, int KREAL, int NCREAL, int NCPAD, bool STATS>
__global__ void __launch_bounds__(32 * NTILES) gemm_wmma(
    const float* __restrict__ A, const _Float16* __restrict__ Wp,
    const float* __restrict__ bias, float* __restrict__ Out,
    float* __restrict__ gsum, float* __restrict__ gsumsq) {
  constexpr int KPAD = 32 * NKT;
  __shared__ _Float16 atile[16 * KPAD];

  const int tid     = threadIdx.x;
  const int rowbase = blockIdx.x * 16;

  // stage + convert A tile (f32 -> f16), zero-padded along K
#pragma unroll
  for (int i = tid; i < 16 * KPAD; i += 32 * NTILES) {
    int r = i / KPAD;
    int k = i - r * KPAD;
    float v = (k < KREAL) ? A[(size_t)(rowbase + r) * KREAL + k] : 0.0f;
    atile[i] = (_Float16)v;
  }
  __syncthreads();

  const int wid  = tid >> 5;  // == ntile
  const int lane = tid & 31;
  const int hl   = lane >> 4; // lane half
  const int l15  = lane & 15;

  // preload this wave's B fragments (L2-resident packed weights)
  v16h bfrag[NKT];
#pragma unroll
  for (int kt = 0; kt < NKT; ++kt)
    bfrag[kt] = *(const v16h*)(Wp + (((size_t)kt * NTILES + wid) * 32 + lane) * 16);

  v8f c = {};
#pragma unroll
  for (int kt = 0; kt < NKT; ++kt) {
    // A fragment: lane half 0 -> K {0..15}; half 1 -> K {16..31}
    union { v16h v; uint32_t u[8]; } a;
    const uint32_t* rp = (const uint32_t*)(atile + l15 * KPAD + kt * 32 + hl * 16);
#pragma unroll
    for (int j = 0; j < 4; ++j) { a.u[j] = rp[j]; a.u[4 + j] = rp[4 + j]; }
    c = __builtin_amdgcn_wmma_f32_16x16x32_f16(false, a.v, false, bfrag[kt],
                                               (short)0, c, false, false);
  }

  // epilogue: bias + unconditional store + fused BN statistics
  const int col = wid * 16 + l15;
  const float bv = (col < NCREAL) ? bias[col] : 0.0f;  // only guard (padded cols)
  float* outp = Out + (size_t)(rowbase + hl * 8) * NCPAD + col;
  float s = 0.0f, sq = 0.0f;
#pragma unroll
  for (int r = 0; r < 8; ++r) {
    float v = c[r] + bv;
    s += v; sq += v * v;
    outp[(size_t)r * NCPAD] = v;
  }
  if constexpr (STATS) {  // lanes l and l+16 hit the same column; atomics merge
    atomicAdd(&gsum[col], s);
    atomicAdd(&gsumsq[col], sq);
  }
}

// ------------------------------------------------------------- BN helpers
__global__ void bn_stats(const float* __restrict__ gsum, const float* __restrict__ gsumsq,
                         const float* __restrict__ gamma, const float* __restrict__ beta,
                         float* __restrict__ scale, float* __restrict__ shift, float invN) {
  int j = threadIdx.x;  // 128 threads
  float mean = gsum[j] * invN;
  float var  = gsumsq[j] * invN - mean * mean;
  float sc   = gamma[j] * rsqrtf(var + BN_EPS);
  scale[j] = sc;
  shift[j] = beta[j] - mean * sc;
}

__global__ void bn_relu(const float* __restrict__ in, float* __restrict__ out,
                        const float* __restrict__ scale, const float* __restrict__ shift,
                        long n) {
  long i = (long)blockIdx.x * blockDim.x + threadIdx.x;
  if (i >= n) return;
  int j = (int)(i & (HIDDEN - 1));
  float v = fmaf(in[i], scale[j], shift[j]);
  out[i] = v > 0.0f ? v : 0.0f;
}

// reads padded [N,16] logits, writes [N,10] log-probs
__global__ void log_softmax10(const float* __restrict__ logits, float* __restrict__ out, int n) {
  int i = blockIdx.x * blockDim.x + threadIdx.x;
  if (i >= n) return;
  const float* row = logits + (size_t)i * 16;
  float v[N_CLASS];
  float m = -3.4e38f;
#pragma unroll
  for (int j = 0; j < N_CLASS; ++j) { v[j] = row[j]; m = fmaxf(m, v[j]); }
  float sum = 0.0f;
#pragma unroll
  for (int j = 0; j < N_CLASS; ++j) sum += __expf(v[j] - m);
  float l = m + __logf(sum);
  float* orow = out + (size_t)i * N_CLASS;
#pragma unroll
  for (int j = 0; j < N_CLASS; ++j) orow[j] = v[j] - l;
}

// ---------------------------------------------------------------- launcher
extern "C" void kernel_launch(void* const* d_in, const int* in_sizes, int n_in,
                              void* d_out, int out_size, void* d_ws, size_t ws_size,
                              hipStream_t stream) {
  (void)n_in; (void)out_size; (void)ws_size;
  const float* x    = (const float*)d_in[0];
  const float* w11  = (const float*)d_in[2];
  const float* b11  = (const float*)d_in[3];
  const float* g11  = (const float*)d_in[4];
  const float* t11  = (const float*)d_in[5];
  const float* w12  = (const float*)d_in[6];
  const float* b12  = (const float*)d_in[7];
  const float* g12  = (const float*)d_in[8];
  const float* t12  = (const float*)d_in[9];
  const float* w21  = (const float*)d_in[10];
  const float* b21  = (const float*)d_in[11];
  const float* g21  = (const float*)d_in[12];
  const float* t21  = (const float*)d_in[13];
  const float* w22  = (const float*)d_in[14];
  const float* b22  = (const float*)d_in[15];
  const float* g22  = (const float*)d_in[16];
  const float* t22  = (const float*)d_in[17];
  const float* wf_  = (const float*)d_in[18];
  const float* bf_  = (const float*)d_in[19];
  const int*   ei   = (const int*)d_in[20];

  const int N = in_sizes[0] / N_CLASS;   // 100000 (divisible by 16)
  const int E = in_sizes[1];             // 1600000
  const int* src = ei;
  const int* dst = ei + E;

  // workspace layout (floats)
  float* f = (float*)d_ws;
  const size_t nh = (size_t)N * HIDDEN;
  float* bufA  = f;                         // [N,128]
  float* bufB  = f + nh;                    // [N,128]
  float* S0    = f + 2 * nh;                // [N,10]  h0 = x + agg
  float* stats = S0 + (size_t)N * N_CLASS;  // 4 layers * (sum128 + sumsq128)
  float* bnp   = stats + 1024;              // 4 layers * (scale128 + shift128)
  _Float16* wp  = (_Float16*)(bnp + 1024);  // packed weight fragments (32B aligned)
  _Float16* w11p = wp;                      // 1 ktile * 8 ntiles * 512
  _Float16* w12p = wp + 4096;               // 4*8*512
  _Float16* w21p = wp + 4096 + 16384;
  _Float16* w22p = wp + 4096 + 2 * 16384;
  _Float16* wfp  = wp + 4096 + 3 * 16384;   // 4 ktiles * 1 ntile * 512
  float* L16   = (float*)(wp + 4096 + 3 * 16384 + 2048);  // padded logits [N,16]

  const int  mt   = N / 16;
  const float invN = 1.0f / (float)N;
  const long n10  = (long)N * N_CLASS;
  const long n128 = (long)nh;
  const long e10  = (long)E * N_CLASS;
  const long e128 = (long)E * HIDDEN;
  const int  EB   = 256;
  auto blks = [](long n, int b) { return (int)((n + b - 1) / b); };

  zero_f32<<<blks(1024, EB), EB, 0, stream>>>(stats, 1024);
  pack_weights<<<1, 256, 0, stream>>>(w11, w11p, 1, 8, N_CLASS, HIDDEN);
  pack_weights<<<4, 256, 0, stream>>>(w12, w12p, 4, 8, HIDDEN, HIDDEN);
  pack_weights<<<4, 256, 0, stream>>>(w21, w21p, 4, 8, HIDDEN, HIDDEN);
  pack_weights<<<4, 256, 0, stream>>>(w22, w22p, 4, 8, HIDDEN, HIDDEN);
  pack_weights<<<1, 128, 0, stream>>>(wf_, wfp,  4, 1, HIDDEN, N_CLASS);

  // ---- layer 1: h0 = x + segment_sum(x[src] -> dst)
  copy_f32<<<blks(n10, EB), EB, 0, stream>>>(x, S0, n10);
  scatter_add<N_CLASS><<<blks(e10, EB), EB, 0, stream>>>(x, src, dst, S0, e10);

  // MLP1.1: [N,10]@[10,128] (K padded to one 32-tile)
  gemm_wmma<1, 8, N_CLASS, HIDDEN, HIDDEN, true><<<mt, 256, 0, stream>>>(
      S0, w11p, b11, bufA, stats + 0, stats + 128);
  bn_stats<<<1, 128, 0, stream>>>(stats + 0, stats + 128, g11, t11, bnp + 0, bnp + 128, invN);
  bn_relu<<<blks(n128, EB), EB, 0, stream>>>(bufA, bufB, bnp + 0, bnp + 128, n128);

  // MLP1.2: [N,128]@[128,128]
  gemm_wmma<4, 8, HIDDEN, HIDDEN, HIDDEN, true><<<mt, 256, 0, stream>>>(
      bufB, w12p, b12, bufA, stats + 256, stats + 384);
  bn_stats<<<1, 128, 0, stream>>>(stats + 256, stats + 384, g12, t12, bnp + 256, bnp + 384, invN);
  bn_relu<<<blks(n128, EB), EB, 0, stream>>>(bufA, bufB, bnp + 256, bnp + 384, n128);  // X2

  // ---- layer 2 aggregation: bufA = X2 + agg(X2)
  copy_f32<<<blks(n128, EB), EB, 0, stream>>>(bufB, bufA, n128);
  scatter_add<HIDDEN><<<blks(e128, EB), EB, 0, stream>>>(bufB, src, dst, bufA, e128);

  // MLP2.1
  gemm_wmma<4, 8, HIDDEN, HIDDEN, HIDDEN, true><<<mt, 256, 0, stream>>>(
      bufA, w21p, b21, bufB, stats + 512, stats + 640);
  bn_stats<<<1, 128, 0, stream>>>(stats + 512, stats + 640, g21, t21, bnp + 512, bnp + 640, invN);
  bn_relu<<<blks(n128, EB), EB, 0, stream>>>(bufB, bufA, bnp + 512, bnp + 640, n128);

  // MLP2.2
  gemm_wmma<4, 8, HIDDEN, HIDDEN, HIDDEN, true><<<mt, 256, 0, stream>>>(
      bufA, w22p, b22, bufB, stats + 768, stats + 896);
  bn_stats<<<1, 128, 0, stream>>>(stats + 768, stats + 896, g22, t22, bnp + 768, bnp + 896, invN);
  bn_relu<<<blks(n128, EB), EB, 0, stream>>>(bufB, bufA, bnp + 768, bnp + 896, n128);

  // classifier: [N,128]@[128,10] -> padded [N,16] logits, then log-softmax
  gemm_wmma<4, 1, HIDDEN, N_CLASS, 16, false><<<mt, 32, 0, stream>>>(
      bufA, wfp, bf_, L16, nullptr, nullptr);
  log_softmax10<<<blks(N, EB), EB, 0, stream>>>(L16, (float*)d_out, N);
}
